// ContrastiveLoss_45200235823668
// MI455X (gfx1250) — matrix operand order repaired and checked by reading, not compiled, not run on hardware
//
#include <hip/hip_runtime.h>
#include <hip/hip_bf16.h>

// ---------------------------------------------------------------------------
// Contrastive loss: sim = normalize(X) @ normalize(X)^T  (4096x4096 from 4096x512)
//   pos_loss = mean_{triu same-label} (1 - sim)^2
//   neg_loss = mean_{diff-label, full} relu(sim - 1)^2
// Strategy: f16 WMMA (f32 accumulate) Gram tiles, upper-triangular macro-tiles
// only, fused masked reduction (sim never materialized).
// ---------------------------------------------------------------------------

typedef __attribute__((ext_vector_type(16))) _Float16 v16h;
typedef __attribute__((ext_vector_type(8)))  _Float16 v8h_t;
typedef __attribute__((ext_vector_type(8)))  float    v8f;

#define NROWS 4096
#define NDIM  512
#define TB    128            // macro-tile grid dim: 4096/32

// ---------------------------------------------------------------------------
// Kernel 1: row L2-normalize fp32 -> f16, row-major
// ---------------------------------------------------------------------------
__global__ __launch_bounds__(256)
void normalize_f16_kernel(const float* __restrict__ X, _Float16* __restrict__ Xn) {
    __shared__ float red[256];
    const int row = blockIdx.x;
    const int t   = threadIdx.x;
    const float* xr = X + (size_t)row * NDIM;
    float x0 = xr[t];
    float x1 = xr[t + 256];
    red[t] = x0 * x0 + x1 * x1;
    __syncthreads();
    for (int s = 128; s > 0; s >>= 1) {
        if (t < s) red[t] += red[t + s];
        __syncthreads();
    }
    const float scale = 1.0f / fmaxf(sqrtf(red[0]), 1e-8f);
    _Float16* o = Xn + (size_t)row * NDIM;
    o[t]       = (_Float16)(x0 * scale);
    o[t + 256] = (_Float16)(x1 * scale);
}

// ---------------------------------------------------------------------------
// WMMA helpers
// ---------------------------------------------------------------------------
// A-operand (16x32 f16), K contiguous to match B: lane l (l<16): M=l, K 0..15;
// lanes 16-31: M=l-16, K 16..31.
// => per lane: two contiguous 8-halfword (16B) chunks at base and base+8.
__device__ __forceinline__ v16h load_a_frag(const _Float16* __restrict__ p, int base) {
    v8h_t lo = *(const v8h_t*)(p + base);
    v8h_t hi = *(const v8h_t*)(p + base + 8);
    v16h r;
#pragma unroll
    for (int i = 0; i < 8; ++i) { r[i] = lo[i]; r[8 + i] = hi[i]; }
    return r;
}

__device__ __forceinline__ v8f wmma_f16(v16h a, v16h b, v8f c) {
    return __builtin_amdgcn_wmma_f32_16x16x32_f16(
        /*neg_a=*/false, a, /*neg_b=*/false, b,
        /*c_mod=*/(short)0, c, /*reuse_a=*/false, /*reuse_b=*/false);
}

// Per-element masked accumulation matching the reference.
__device__ __forceinline__ void acc_elem(float s, int i, int j, int li, int lj,
                                         float wneg, float& pos, float& neg) {
    if (li == lj) {
        if (i < j) { float d = 1.0f - s; pos += d * d; }
    } else {
        float t = s - 1.0f;               // MARGIN = 1.0
        if (t > 0.0f) neg += wneg * t * t;
    }
}

// ---------------------------------------------------------------------------
// Kernel 2: one wave per 32x32 macro-tile of sim; upper triangle (bi<=bj) only.
// 4 WMMA accumulators (2x2 of 16x16), K-loop over 512 in steps of 32.
// ---------------------------------------------------------------------------
__global__ __launch_bounds__(32)
void gram_tile_kernel(const _Float16* __restrict__ Xn,
                      const int* __restrict__ targets,
                      float2* __restrict__ partials) {
    const int bj   = blockIdx.x;
    const int bi   = blockIdx.y;
    const int bidx = bi * TB + bj;
    const int l    = threadIdx.x;        // 0..31, wave32
    if (bj < bi) {                       // mirrored macro-tile: nothing to do
        if (l == 0) partials[bidx] = make_float2(0.0f, 0.0f);
        return;
    }
    const int lm   = l & 15;
    const int half = l >> 4;
    const int i0   = bi * 32;
    const int j0   = bj * 32;

    const _Float16* a0p = Xn + (size_t)(i0 + lm)      * NDIM;
    const _Float16* a1p = Xn + (size_t)(i0 + 16 + lm) * NDIM;
    const _Float16* b0p = Xn + (size_t)(j0 + lm)      * NDIM;
    const _Float16* b1p = Xn + (size_t)(j0 + 16 + lm) * NDIM;

    v8f c00 = {}, c01 = {}, c10 = {}, c11 = {};

#pragma unroll 4
    for (int k0 = 0; k0 < NDIM; k0 += 32) {
        const int abase = k0 + 16 * half;         // A chunk base per lane-half
        v16h a0 = load_a_frag(a0p, abase);
        v16h a1 = load_a_frag(a1p, abase);
        // B-operand (32x16): lane = N (mod 16), lanes 16-31 take K+16,
        // K contiguous across VGPRs => one 32B contiguous load per lane.
        v16h b0 = *(const v16h*)(b0p + k0 + 16 * half);
        v16h b1 = *(const v16h*)(b1p + k0 + 16 * half);
        c00 = wmma_f16(a0, b0, c00);
        c01 = wmma_f16(a0, b1, c01);
        c10 = wmma_f16(a1, b0, c10);
        c11 = wmma_f16(a1, b1, c11);
    }

    // Epilogue: C/D layout => lane holds column j = j0 + lm (+16 for *1 tiles),
    // VGPR r holds row M = r + 8*half (+16 for 1* tiles).
    const float wneg = (bi == bj) ? 1.0f : 2.0f;  // mirror of off-diag tiles
    const int jA = j0 + lm,       jB = j0 + 16 + lm;
    const int labjA = targets[jA], labjB = targets[jB];

    float pos = 0.0f, neg = 0.0f;
#pragma unroll
    for (int r = 0; r < 8; ++r) {
        const int iA = i0 + r + 8 * half;
        const int iB = iA + 16;
        const int labiA = targets[iA];
        const int labiB = targets[iB];
        acc_elem(c00[r], iA, jA, labiA, labjA, wneg, pos, neg);
        acc_elem(c01[r], iA, jB, labiA, labjB, wneg, pos, neg);
        acc_elem(c10[r], iB, jA, labiB, labjA, wneg, pos, neg);
        acc_elem(c11[r], iB, jB, labiB, labjB, wneg, pos, neg);
    }

    // Deterministic wave reduction, one float2 partial per macro-tile.
#pragma unroll
    for (int off = 16; off > 0; off >>= 1) {
        pos += __shfl_xor(pos, off, 32);
        neg += __shfl_xor(neg, off, 32);
    }
    if (l == 0) partials[bidx] = make_float2(pos, neg);
}

// ---------------------------------------------------------------------------
// Kernel 3: label histogram -> exact pair counts; reduce partials; emit loss.
// ---------------------------------------------------------------------------
__global__ __launch_bounds__(256)
void finalize_kernel(const int* __restrict__ targets, int n,
                     const float2* __restrict__ partials, int np,
                     float* __restrict__ out) {
    __shared__ int   hist[64];
    __shared__ float sp[256];
    __shared__ float sn[256];
    const int t = threadIdx.x;
    if (t < 64) hist[t] = 0;
    __syncthreads();
    for (int i = t; i < n; i += 256) atomicAdd(&hist[targets[i] & 63], 1);
    __syncthreads();

    float p = 0.0f, ng = 0.0f;
    for (int i = t; i < np; i += 256) {
        float2 v = partials[i];
        p  += v.x;
        ng += v.y;
    }
    sp[t] = p; sn[t] = ng;
    __syncthreads();
    for (int s = 128; s > 0; s >>= 1) {
        if (t < s) { sp[t] += sp[t + s]; sn[t] += sn[t + s]; }
        __syncthreads();
    }
    if (t == 0) {
        long long posc = 0;
        long long negc = (long long)n * (long long)n;
        for (int b = 0; b < 64; ++b) {
            long long c = hist[b];
            posc += c * (c - 1) / 2;
            negc -= c * c;
        }
        float pl = (posc > 0) ? sp[0] / (float)posc : 0.0f;
        float nl = (negc > 0) ? sn[0] / (float)negc : 0.0f;
        out[0] = pl + nl;
    }
}

// ---------------------------------------------------------------------------
extern "C" void kernel_launch(void* const* d_in, const int* in_sizes, int n_in,
                              void* d_out, int out_size, void* d_ws, size_t ws_size,
                              hipStream_t stream) {
    const float* X       = (const float*)d_in[0];   // [4096, 512] fp32
    const int*   targets = (const int*)d_in[1];     // [4096] int
    float*       out     = (float*)d_out;           // scalar fp32

    _Float16* Xn = (_Float16*)d_ws;                                    // 4 MB
    float2* partials = (float2*)((char*)d_ws +
                        (size_t)NROWS * NDIM * sizeof(_Float16));      // 128 KB

    normalize_f16_kernel<<<NROWS, 256, 0, stream>>>(X, Xn);

    dim3 grid(TB, TB);
    gram_tile_kernel<<<grid, 32, 0, stream>>>(Xn, targets, partials);

    finalize_kernel<<<1, 256, 0, stream>>>(targets, NROWS, partials, TB * TB, out);
}
